// LSTM_51926154609261
// MI455X (gfx1250) — matrix operand order repaired and checked by reading, not compiled, not run on hardware
//
#include <hip/hip_runtime.h>
#include <hip/hip_bf16.h>

// ---------------------------------------------------------------------------
// LSTM (T=256, B=64, D=H=1024) via weight folding + bf16 WMMA on gfx1250.
//   Ax = Wx_g @ W_in   [4H x D]   (one-time fold, bf16)
//   Ah = Wh_g @ W_hid  [4H x H]   (one-time fold, bf16)
//   bias = Wx_g@b_in + bx_g + Wh_g@b_hid + bh_g   [4H]
//   GX  = X @ Ax^T     [T*B x 4H] (one big parallel GEMM, bf16 out)
//   per step: G = h @ Ah^T + GX[t] + bias ; gates ; c,h update
// Matrix math: v_wmma_f32_16x16x32_bf16 (f32 accumulate).
// Shared A tiles staged block-wide with global_load_async_to_lds_b128
// (ASYNCcnt) and consumed via ds_load_b128.
// ---------------------------------------------------------------------------

typedef __attribute__((ext_vector_type(16))) __bf16       v16bf;
typedef __attribute__((ext_vector_type(8)))  float        v8f;
typedef __attribute__((ext_vector_type(8)))  unsigned int v8u;
typedef __attribute__((ext_vector_type(4)))  unsigned int v4u;
typedef __attribute__((ext_vector_type(4)))  float        v4f;

#define Tn 256
#define Bn 64
#define Dn 1024
#define Hn 1024
#define G4 4096   // 4*H

#define WMMA_BF16(A_, B_, C_) \
  __builtin_amdgcn_wmma_f32_16x16x32_bf16(false, (A_), false, (B_), (short)0, (C_), false, false)

__device__ __forceinline__ v16bf pack2(v4u lo, v4u hi) {
  v8u u;
  u[0]=lo[0]; u[1]=lo[1]; u[2]=lo[2]; u[3]=lo[3];
  u[4]=hi[0]; u[5]=hi[1]; u[6]=hi[2]; u[7]=hi[3];
  return __builtin_bit_cast(v16bf, u);
}

// Async DMA: one b128 from global -> LDS per active lane (tracked by ASYNCcnt).
__device__ __forceinline__ void async_b128_to_lds(const void* gptr, void* lptr) {
  unsigned lds_off = (unsigned)(size_t)lptr;   // low 32 bits of flat LDS addr = LDS offset
  asm volatile("global_load_async_to_lds_b128 %0, %1, off"
               :: "v"(lds_off), "v"(gptr) : "memory");
}
__device__ __forceinline__ void wait_async_and_sync() {
  asm volatile("s_wait_asynccnt 0" ::: "memory");
  __syncthreads();
}

// A-fragment (16xK tile row slice) from bf16 row-major [rows][ld].
// Layout used here (matches b_frag_wt): lane L -> M=L%16; elems 0..15: K=k0+16*sub .. +15 (contiguous in k)
__device__ __forceinline__ v16bf a_frag_bf16(const __bf16* base, int ld, int m, int k0, int sub) {
  const __bf16* p = base + (size_t)m * ld + k0 + sub * 16;
  v4u lo = *(const v4u*)p;          // 8 bf16, K = k0+16*sub .. +7
  v4u hi = *(const v4u*)(p + 8);    // 8 bf16, K = k0+16*sub+8 .. +15
  return pack2(lo, hi);
}

// Same A-fragment but sourced from f32 with on-the-fly bf16 convert.
__device__ __forceinline__ v16bf a_frag_f32(const float* base, int ld, int m, int k0, int sub) {
  const float* p = base + (size_t)m * ld + k0 + sub * 16;
  v4f f0 = *(const v4f*)(p);      v4f f1 = *(const v4f*)(p + 4);
  v4f f2 = *(const v4f*)(p + 8);  v4f f3 = *(const v4f*)(p + 12);
  v16bf a;
  a[0]=(__bf16)f0[0];  a[1]=(__bf16)f0[1];  a[2]=(__bf16)f0[2];  a[3]=(__bf16)f0[3];
  a[4]=(__bf16)f1[0];  a[5]=(__bf16)f1[1];  a[6]=(__bf16)f1[2];  a[7]=(__bf16)f1[3];
  a[8]=(__bf16)f2[0];  a[9]=(__bf16)f2[1];  a[10]=(__bf16)f2[2]; a[11]=(__bf16)f2[3];
  a[12]=(__bf16)f3[0]; a[13]=(__bf16)f3[1]; a[14]=(__bf16)f3[2]; a[15]=(__bf16)f3[3];
  return a;
}

// B-fragment for C = A * W^T, W row-major bf16 [N][K]: B[k][n] = W[n][k].
// Layout used here (same as A frags): lane L -> N=L%16; elems 0..15: K = k0+16*sub .. +15 (contiguous in k!).
__device__ __forceinline__ v16bf b_frag_wt(const __bf16* w, int ld, int n, int k0, int sub) {
  const __bf16* p = w + (size_t)n * ld + k0 + sub * 16;
  v4u lo = *(const v4u*)p;
  v4u hi = *(const v4u*)(p + 8);
  return pack2(lo, hi);
}

// ---------------------------------------------------------------------------
// K1: one-time weight fold  C[M][N](bf16) = A[M][K](f32) @ B[K][N](f32)
//     one 16x16 tile per wave; B gathered strided (one-time cost only).
// ---------------------------------------------------------------------------
__global__ __launch_bounds__(256) void fold_gemm_nn(const float* __restrict__ A,
                                                    const float* __restrict__ B,
                                                    __bf16* __restrict__ C,
                                                    int M, int N, int K) {
  int wid  = blockIdx.x * 8 + (threadIdx.x >> 5);
  int lane = threadIdx.x & 31;
  int sub  = lane >> 4, lc = lane & 15;
  int ntn  = N >> 4;
  int mT   = (wid / ntn) << 4;
  int nT   = (wid % ntn) << 4;
  int m    = mT + lc;
  v8f acc = {};
  for (int k0 = 0; k0 < K; k0 += 32) {
    v16bf a = a_frag_f32(A, K, m, k0, sub);
    v16bf b;
    const float* bp = B + (size_t)(k0 + sub * 16) * N + nT + lc;
#pragma unroll
    for (int e = 0; e < 16; ++e) b[e] = (__bf16)bp[(size_t)e * N];
    acc = WMMA_BF16(a, b, acc);
  }
#pragma unroll
  for (int r = 0; r < 8; ++r) {
    int row = mT + r + 8 * sub;
    C[(size_t)row * N + nT + lc] = (__bf16)acc[r];
  }
}

// ---------------------------------------------------------------------------
// K2: bias fold  bias[j] = Wx[j,:]·b_in + bx[j] + Wh[j,:]·b_hid + bh[j]
// ---------------------------------------------------------------------------
__global__ void fold_bias(const float* __restrict__ Wx, const float* __restrict__ bin,
                          const float* __restrict__ bx, const float* __restrict__ Wh,
                          const float* __restrict__ bh_in, const float* __restrict__ bh,
                          float* __restrict__ bias) {
  int j = blockIdx.x * 256 + threadIdx.x;   // 0..4095
  const float* wx = Wx + (size_t)j * Hn;
  const float* wh = Wh + (size_t)j * Hn;
  float acc = bx[j] + bh[j];
  for (int k = 0; k < Hn; ++k) acc += wx[k] * bin[k] + wh[k] * bh_in[k];
  bias[j] = acc;
}

// ---------------------------------------------------------------------------
// K3: state init (f32 + bf16 shadow of h)
// ---------------------------------------------------------------------------
__global__ void init_state(const float* __restrict__ h0, const float* __restrict__ c0,
                           float* __restrict__ h, float* __restrict__ c,
                           __bf16* __restrict__ hbf) {
  int i = blockIdx.x * 256 + threadIdx.x;   // 0..65535
  float hv = h0[i];
  h[i] = hv; hbf[i] = (__bf16)hv; c[i] = c0[i];
}

// ---------------------------------------------------------------------------
// K4: GX = X[16384x1024](f32) @ Ax^T (Ax bf16 row-major [4096][1024]) -> bf16
//     All 8 waves of a block share one 16-row A tile: async-DMA it to LDS
//     (64 KB f32) once, then each wave feeds WMMA from ds_load_b128.
// ---------------------------------------------------------------------------
__global__ __launch_bounds__(256) void gx_gemm(const float* __restrict__ X,
                                               const __bf16* __restrict__ W,
                                               __bf16* __restrict__ GX) {
  __shared__ __align__(16) float sX[16 * Dn];     // 64 KB
  int wid  = blockIdx.x * 8 + (threadIdx.x >> 5);
  int lane = threadIdx.x & 31;
  int sub  = lane >> 4, lc = lane & 15;
  int mT = (wid >> 6) << 4;        // 1024 m-tiles; uniform within a block
  int nG = (wid & 63) << 6;        // 64 n-groups of 64

  // Cooperative async stage: 16 rows x 1024 f32 = 4096 b128 chunks, 16/thread.
  {
    const float* src = X + (size_t)mT * Dn;
#pragma unroll
    for (int i = 0; i < 16; ++i) {
      int e4 = ((int)threadIdx.x + i * 256) * 4;   // float index, 16B chunks
      async_b128_to_lds(src + e4, &sX[e4]);
    }
    wait_async_and_sync();
  }

  v8f a0 = {}, a1 = {}, a2 = {}, a3 = {};
  for (int k0 = 0; k0 < Dn; k0 += 32) {
    v16bf a  = a_frag_f32(sX, Dn, lc, k0, sub);    // ds_load_b128 path
    v16bf b0 = b_frag_wt(W, Dn, nG + 0  + lc, k0, sub);
    v16bf b1 = b_frag_wt(W, Dn, nG + 16 + lc, k0, sub);
    v16bf b2 = b_frag_wt(W, Dn, nG + 32 + lc, k0, sub);
    v16bf b3 = b_frag_wt(W, Dn, nG + 48 + lc, k0, sub);
    a0 = WMMA_BF16(a, b0, a0);
    a1 = WMMA_BF16(a, b1, a1);
    a2 = WMMA_BF16(a, b2, a2);
    a3 = WMMA_BF16(a, b3, a3);
  }
#pragma unroll
  for (int r = 0; r < 8; ++r) {
    int row = mT + r + 8 * sub;
    size_t g = (size_t)row * G4 + nG + lc;
    GX[g + 0]  = (__bf16)a0[r];
    GX[g + 16] = (__bf16)a1[r];
    GX[g + 32] = (__bf16)a2[r];
    GX[g + 48] = (__bf16)a3[r];
  }
}

// ---------------------------------------------------------------------------
// K5: per-step  G[64][4096](f32) = h_bf16 @ Ah^T + GX[t] + bias
//     Shared 16-row bf16 h tile (32 KB) async-staged to LDS per block.
// ---------------------------------------------------------------------------
__global__ __launch_bounds__(256) void step_gemm(const __bf16* __restrict__ hbf,
                                                 const __bf16* __restrict__ W,
                                                 const __bf16* __restrict__ GXt,
                                                 const float* __restrict__ bias,
                                                 float* __restrict__ G) {
  __shared__ __align__(16) __bf16 sH[16 * Hn];    // 32 KB
  int wid  = blockIdx.x * 8 + (threadIdx.x >> 5);  // 256 waves total
  int lane = threadIdx.x & 31;
  int sub  = lane >> 4, lc = lane & 15;
  int mT = (wid >> 6) << 4;   // 4 m-tiles; uniform within a block
  int nG = (wid & 63) << 6;   // 64 n-groups

  // Cooperative async stage: 16 rows x 1024 bf16 = 2048 b128 chunks, 8/thread.
  {
    const __bf16* src = hbf + (size_t)mT * Hn;
#pragma unroll
    for (int i = 0; i < 8; ++i) {
      int e8 = ((int)threadIdx.x + i * 256) * 8;   // bf16 index, 16B chunks
      async_b128_to_lds(src + e8, &sH[e8]);
    }
    wait_async_and_sync();
  }

  v8f a0 = {}, a1 = {}, a2 = {}, a3 = {};
  for (int k0 = 0; k0 < Hn; k0 += 32) {
    v16bf a  = a_frag_bf16(sH, Hn, lc, k0, sub);   // ds_load_b128 path
    v16bf b0 = b_frag_wt(W, Hn, nG + 0  + lc, k0, sub);
    v16bf b1 = b_frag_wt(W, Hn, nG + 16 + lc, k0, sub);
    v16bf b2 = b_frag_wt(W, Hn, nG + 32 + lc, k0, sub);
    v16bf b3 = b_frag_wt(W, Hn, nG + 48 + lc, k0, sub);
    a0 = WMMA_BF16(a, b0, a0);
    a1 = WMMA_BF16(a, b1, a1);
    a2 = WMMA_BF16(a, b2, a2);
    a3 = WMMA_BF16(a, b3, a3);
  }
#pragma unroll
  for (int r = 0; r < 8; ++r) {
    int row = mT + r + 8 * sub;
    size_t g = (size_t)row * G4 + nG + lc;
    G[g + 0]  = a0[r] + (float)GXt[g + 0]  + bias[nG + 0  + lc];
    G[g + 16] = a1[r] + (float)GXt[g + 16] + bias[nG + 16 + lc];
    G[g + 32] = a2[r] + (float)GXt[g + 32] + bias[nG + 32 + lc];
    G[g + 48] = a3[r] + (float)GXt[g + 48] + bias[nG + 48 + lc];
  }
}

// ---------------------------------------------------------------------------
// K6: per-step gate math + cell/hidden update
// ---------------------------------------------------------------------------
__global__ void gate_update(const float* __restrict__ G, float* __restrict__ c,
                            float* __restrict__ h, __bf16* __restrict__ hbf) {
  int idx = blockIdx.x * 256 + threadIdx.x;   // 0..65535 = b*1024 + j
  int b = idx >> 10, j = idx & 1023;
  const float* gr = G + (size_t)b * G4;
  float gi = gr[j], gf = gr[Hn + j], gc = gr[2 * Hn + j], go = gr[3 * Hn + j];
  float i_ = 1.0f / (1.0f + __expf(-gi));
  float f_ = 1.0f / (1.0f + __expf(-gf));
  float g_ = tanhf(gc);
  float o_ = 1.0f / (1.0f + __expf(-go));
  float cn = f_ * c[idx] + i_ * g_;
  float hn = o_ * tanhf(cn);
  c[idx] = cn; h[idx] = hn; hbf[idx] = (__bf16)hn;
}

// ---------------------------------------------------------------------------
// Workspace layout (bytes). Total ~146 MiB.
// ---------------------------------------------------------------------------
#define OFF_AXB   ((size_t)0)                               // bf16 [4096][1024]  8 MiB
#define OFF_AHB   (OFF_AXB + (size_t)4096 * 1024 * 2)       // bf16 [4096][1024]  8 MiB
#define OFF_BIAS  (OFF_AHB + (size_t)4096 * 1024 * 2)       // f32  [4096]
#define OFF_HF    (OFF_BIAS + (size_t)4096 * 4)             // f32  [64][1024]
#define OFF_CF    (OFF_HF + (size_t)65536 * 4)              // f32  [64][1024]
#define OFF_HBF   (OFF_CF + (size_t)65536 * 4)              // bf16 [64][1024]
#define OFF_GF    (OFF_HBF + (size_t)65536 * 2)             // f32  [64][4096]    1 MiB
#define OFF_GX    (OFF_GF + (size_t)64 * 4096 * 4)          // bf16 [16384][4096] 128 MiB

extern "C" void kernel_launch(void* const* d_in, const int* in_sizes, int n_in,
                              void* d_out, int out_size, void* d_ws, size_t ws_size,
                              hipStream_t stream) {
  (void)in_sizes; (void)n_in; (void)out_size; (void)ws_size;
  const float* x      = (const float*)d_in[0];   // [256,64,1024]
  const float* hidden = (const float*)d_in[1];   // [64,1024]
  const float* cell   = (const float*)d_in[2];   // [64,1024]
  const float* W_in   = (const float*)d_in[3];   // [1024,1024]
  const float* b_in   = (const float*)d_in[4];   // [1024]
  const float* W_hid  = (const float*)d_in[5];   // [1024,1024]
  const float* b_hid  = (const float*)d_in[6];   // [1024]
  const float* Wx_g   = (const float*)d_in[7];   // [4,1024,1024] (flat [4096][1024])
  const float* bx_g   = (const float*)d_in[8];   // [4,1024]
  const float* Wh_g   = (const float*)d_in[9];   // [4,1024,1024]
  const float* bh_g   = (const float*)d_in[10];  // [4,1024]

  char*   ws   = (char*)d_ws;
  __bf16* Axb  = (__bf16*)(ws + OFF_AXB);
  __bf16* Ahb  = (__bf16*)(ws + OFF_AHB);
  float*  bias = (float*)(ws + OFF_BIAS);
  float*  hf   = (float*)(ws + OFF_HF);
  float*  cf   = (float*)(ws + OFF_CF);
  __bf16* hbf  = (__bf16*)(ws + OFF_HBF);
  float*  Gf   = (float*)(ws + OFF_GF);
  __bf16* GX   = (__bf16*)(ws + OFF_GX);

  // One-time folds: Ax = Wx_g @ W_in, Ah = Wh_g @ W_hid  (each M=4096,N=1024,K=1024)
  fold_gemm_nn<<<2048, 256, 0, stream>>>(Wx_g, W_in,  Axb, 4096, 1024, 1024);
  fold_gemm_nn<<<2048, 256, 0, stream>>>(Wh_g, W_hid, Ahb, 4096, 1024, 1024);
  fold_bias<<<16, 256, 0, stream>>>(Wx_g, b_in, bx_g, Wh_g, b_hid, bh_g, bias);
  init_state<<<256, 256, 0, stream>>>(hidden, cell, hf, cf, hbf);

  // All-timestep x-path: GX = X @ Ax^T  (M=16384, N=4096, K=1024)
  gx_gemm<<<8192, 256, 0, stream>>>(x, Axb, GX);

  // Sequential recurrence
  for (int t = 0; t < Tn; ++t) {
    step_gemm<<<32, 256, 0, stream>>>(hbf, Ahb, GX + (size_t)t * Bn * G4, bias, Gf);
    gate_update<<<256, 256, 0, stream>>>(Gf, cf, hf, hbf);
  }

  // Output: (h, c) concatenated, f32
  hipMemcpyAsync(d_out, hf, (size_t)Bn * Hn * 4, hipMemcpyDeviceToDevice, stream);
  hipMemcpyAsync((float*)d_out + Bn * Hn, cf, (size_t)Bn * Hn * 4,
                 hipMemcpyDeviceToDevice, stream);
}
